// QuanvolutionClassifier_65481071399845
// MI455X (gfx1250) — matrix operand order repaired and checked by reading, not compiled, not run on hardware
//
#include <hip/hip_runtime.h>
#include <hip/hip_bf16.h>

typedef _Float16 v16h __attribute__((ext_vector_type(16)));
typedef float    v8f  __attribute__((ext_vector_type(8)));

#define BROWS   16            // batch rows per workgroup (WMMA M)
#define PATCHES 196
#define KPAD    800           // 784 padded to a multiple of 32
#define NKCH    (KPAD / 32)   // 25 K-chunks of the 16x16x32 WMMA

__global__ __launch_bounds__(256)
void quanv_fused_kernel(const float* __restrict__ x,
                        const float* __restrict__ theta,
                        const float* __restrict__ W,
                        const float* __restrict__ bias,
                        float* __restrict__ out)
{
    __shared__ __align__(16) _Float16 sF[BROWS][KPAD]; // f16 features, [row][k]
    __shared__ __align__(16) _Float16 sW[16][KPAD];    // f16 W, rows 10..15 / k>=784 zero
    __shared__ float sC[16][16];                       // WMMA result spill

    const int tid = threadIdx.x;
    const int m0  = blockIdx.x * BROWS;

    // ---- stage W into LDS as f16, zero-padded to 16 x 800 ----
    for (int i = tid; i < 16 * KPAD; i += 256) {
        int n = i / KPAD, k = i - n * KPAD;
        float w = (n < 10 && k < 784) ? W[n * 784 + k] : 0.0f;
        sW[n][k] = (_Float16)w;
    }
    // zero the K-tail of the feature tile (cols 784..799, 16x16 = 256 elems)
    {
        int r = tid >> 4, k = 784 + (tid & 15);
        sF[r][k] = (_Float16)0.0f;
    }

    // per-thread learned-rotation constants (uniform; 8 trig ops, hoisted)
    float ct[4], st[4];
    #pragma unroll
    for (int k = 0; k < 4; ++k)
        __sincosf(theta[k] * 0.5f, &st[k], &ct[k]);

    // ---- stage 1: 4-qubit state simulation, one (row, patch) per task ----
    for (int t = tid; t < BROWS * PATCHES; t += 256) {
        int r  = t / PATCHES;
        int p  = t - r * PATCHES;
        int pr = p / 14, pc = p - pr * 14;
        const float* xb = x + (size_t)(m0 + r) * 784 + pr * 56 + pc * 2;
        float2 top = *(const float2*)xb;          // x[2pr, 2pc], x[2pr, 2pc+1]
        float2 bot = *(const float2*)(xb + 28);   // x[2pr+1, 2pc], x[2pr+1, 2pc+1]

        float c[4], s[4];
        __sincosf(top.x * 0.5f, &s[0], &c[0]);
        __sincosf(top.y * 0.5f, &s[1], &c[1]);
        __sincosf(bot.x * 0.5f, &s[2], &c[2]);
        __sincosf(bot.y * 0.5f, &s[3], &c[3]);

        // product state, idx = w0*8 + w1*4 + w2*2 + w3
        float st16[16];
        #pragma unroll
        for (int i = 0; i < 16; ++i) {
            float a0 = (i & 8) ? s[0] : c[0];
            float a1 = (i & 4) ? s[1] : c[1];
            float a2 = (i & 2) ? s[2] : c[2];
            float a3 = (i & 1) ? s[3] : c[3];
            st16[i] = a0 * a1 * a2 * a3;
        }

        #pragma unroll
        for (int d = 0; d < 2; ++d) {
            // RY(theta_k) on wire k (bit 8>>k)
            #pragma unroll
            for (int k = 0; k < 4; ++k) {
                int bit = 8 >> k;
                float cc = ct[k], ss = st[k];
                #pragma unroll
                for (int i = 0; i < 16; ++i) {
                    if (!(i & bit)) {
                        float a0 = st16[i], a1 = st16[i | bit];
                        st16[i]       = cc * a0 - ss * a1;
                        st16[i | bit] = ss * a0 + cc * a1;
                    }
                }
            }
            // CNOT(0,1): where w0==1 flip w1 -> swap 8+j <-> 12+j
            #pragma unroll
            for (int j = 0; j < 4; ++j) {
                float tmp = st16[8 + j]; st16[8 + j] = st16[12 + j]; st16[12 + j] = tmp;
            }
            // CNOT(2,3): where w2==1 flip w3 -> swap b+2 <-> b+3
            #pragma unroll
            for (int b = 0; b < 16; b += 4) {
                float tmp = st16[b + 2]; st16[b + 2] = st16[b + 3]; st16[b + 3] = tmp;
            }
        }

        float z0 = 0.f, z1 = 0.f, z2 = 0.f, z3 = 0.f;
        #pragma unroll
        for (int i = 0; i < 16; ++i) {
            float pi = st16[i] * st16[i];
            z0 += (i & 8) ? -pi : pi;
            z1 += (i & 4) ? -pi : pi;
            z2 += (i & 2) ? -pi : pi;
            z3 += (i & 1) ? -pi : pi;
        }
        _Float16* dst = &sF[r][4 * p];
        dst[0] = (_Float16)z0;
        dst[1] = (_Float16)z1;
        dst[2] = (_Float16)z2;
        dst[3] = (_Float16)z3;
    }

    __syncthreads();

    // ---- stage 2: 16x16 projection via V_WMMA_F32_16X16X32_F16, wave 0 ----
    if (tid < 32) {
        const int lane = tid;
        const int hi   = (lane >= 16);
        const int arow = lane & 15;       // A row (M), both half-waves
        const int bcol = lane & 15;       // B col (N)
        const int asel = hi ? 8  : 0;     // A: lanes>=16 hold K+8 / K+24 runs
        const int bsel = hi ? 8  : 0;     // B: lanes>=16 hold K+8 / K+24 runs

        v8f acc = {0.f, 0.f, 0.f, 0.f, 0.f, 0.f, 0.f, 0.f};
        #pragma unroll
        for (int cch = 0; cch < NKCH; ++cch) {
            int kk = cch * 32;
            union { v16h v; uint4 q[2]; } a, b;
            // A 16x32 f16 fragment: K runs [kk+asel, +8) and [kk+16+asel, +8)
            a.q[0] = *(const uint4*)&sF[arow][kk + asel];
            a.q[1] = *(const uint4*)&sF[arow][kk + 16 + asel];
            // B 32x16 f16 fragment: B[k][n] = W[n][k] -> K runs [kk+bsel, +8) and [kk+16+bsel, +8) of row n
            b.q[0] = *(const uint4*)&sW[bcol][kk + bsel];
            b.q[1] = *(const uint4*)&sW[bcol][kk + 16 + bsel];
            acc = __builtin_amdgcn_wmma_f32_16x16x32_f16(
                      /*neg_a=*/false, a.v, /*neg_b=*/false, b.v,
                      /*c_mod=*/(short)0, acc, /*reuse_a=*/false, /*reuse_b=*/false);
        }
        // C layout: VGPR j -> row j + 8*(lane>=16), col lane%16
        #pragma unroll
        for (int j = 0; j < 8; ++j)
            sC[j + (hi ? 8 : 0)][lane & 15] = acc[j];
    }

    __syncthreads();

    // ---- stage 3: bias + log_softmax over the 10 valid columns ----
    if (tid < 16) {
        float l[10];
        float mx = -1e30f;
        #pragma unroll
        for (int n = 0; n < 10; ++n) {
            l[n] = sC[tid][n] + bias[n];
            mx = fmaxf(mx, l[n]);
        }
        float se = 0.f;
        #pragma unroll
        for (int n = 0; n < 10; ++n) se += __expf(l[n] - mx);
        float lse = mx + __logf(se);
        float* o = out + (size_t)(m0 + tid) * 10;
        #pragma unroll
        for (int n = 0; n < 10; ++n) o[n] = l[n] - lse;
    }
}

extern "C" void kernel_launch(void* const* d_in, const int* in_sizes, int n_in,
                              void* d_out, int out_size, void* d_ws, size_t ws_size,
                              hipStream_t stream) {
    (void)n_in; (void)out_size; (void)d_ws; (void)ws_size;
    const float* x     = (const float*)d_in[0];   // [B, 784]
    const float* theta = (const float*)d_in[1];   // [4]
    const float* W     = (const float*)d_in[2];   // [10, 784]
    const float* bias  = (const float*)d_in[3];   // [10]
    float* out = (float*)d_out;                   // [B, 10]

    const int B = in_sizes[0] / 784;              // 8192
    dim3 grid(B / BROWS);                         // 512 workgroups
    quanv_fused_kernel<<<grid, 256, 0, stream>>>(x, theta, W, bias, out);
}
